// MultiHeadAttention_23364622090567
// MI455X (gfx1250) — matrix operand order repaired and checked by reading, not compiled, not run on hardware
//
#include <hip/hip_runtime.h>
#include <hip/hip_bf16.h>

typedef __attribute__((ext_vector_type(16))) _Float16 v16h;
typedef __attribute__((ext_vector_type(8)))  _Float16 v8h;
typedef __attribute__((ext_vector_type(4)))  _Float16 v4h;
typedef __attribute__((ext_vector_type(8)))  float    v8f;

#define S_LEN 4096
#define D_MODEL 1024
#define NHEAD 16
#define DHEAD 64
#define PH 4128   // f16 score pitch (halves): 4128*2/4 = 2064 words, 2064%64==16
                  // -> conflict-free strided softmax; 16B-aligned rows for b128

static __device__ __forceinline__ void wmma_f16(v16h a, v16h b, v8f& c) {
    c = __builtin_amdgcn_wmma_f32_16x16x32_f16(false, a, false, b, (short)0, c, false, false);
}

// A-fragment: two contiguous 8-half (16B) runs at p and p+8
static __device__ __forceinline__ v16h load_a16(const _Float16* p) {
    v8h lo = *(const v8h*)p;
    v8h hi = *(const v8h*)(p + 8);
    return __builtin_shufflevector(lo, hi, 0, 1, 2, 3, 4, 5, 6, 7,
                                           8, 9, 10, 11, 12, 13, 14, 15);
}

// ---------------------------------------------------------------------------
// Prep: elementwise f32 -> f16 (activations). n = 4096*1024, grid 4096 x 256.
// ---------------------------------------------------------------------------
__global__ void __launch_bounds__(256) cvt16(const float* __restrict__ x,
                                             _Float16* __restrict__ o)
{
    const size_t i = ((size_t)blockIdx.x * 256 + threadIdx.x) * 4;
    const float4 v = *(const float4*)(x + i);
    v4h h;
    h[0] = (_Float16)v.x; h[1] = (_Float16)v.y;
    h[2] = (_Float16)v.z; h[3] = (_Float16)v.w;
    *(v4h*)(o + i) = h;
}

// ---------------------------------------------------------------------------
// Prep: W(f32, k-major 1024x1024) -> WT(f16, n-major). 64x64 tile per block.
// ---------------------------------------------------------------------------
__global__ void __launch_bounds__(256) wt16(const float* __restrict__ W,
                                            _Float16* __restrict__ WT)
{
    __shared__ float t[64][65];
    const int k0 = blockIdx.x * 64;
    const int n0 = blockIdx.y * 64;
    const int r  = threadIdx.x >> 2;
    const int cb = (threadIdx.x & 3) * 16;
    #pragma unroll
    for (int j = 0; j < 16; ++j)
        t[r][cb + j] = W[(size_t)(k0 + r) * D_MODEL + n0 + cb + j];
    __syncthreads();
    #pragma unroll
    for (int j = 0; j < 16; ++j)
        WT[(size_t)(n0 + r) * D_MODEL + k0 + cb + j] = (_Float16)t[cb + j][r];
}

// ---------------------------------------------------------------------------
// LDS-free GEMM: A16(f16, 4096x1024 m-major) @ BT(f16, 1024x1024 n-major) + bias
// Block = 128 threads = 4 waves; wave owns 32(m) x 64(n); all operands are
// direct global vector loads (weights/activations L2-resident).
// vmode 0: out f16 head-major [h][s][64];  vmode 1: out f16 transposed [h][dv][s]
// ---------------------------------------------------------------------------
__global__ void __launch_bounds__(128) gemm16(const _Float16* __restrict__ A16,
                                              const _Float16* __restrict__ BT,
                                              const float* __restrict__ bias,
                                              _Float16* __restrict__ outH,
                                              const int vmode)
{
    const int m0  = blockIdx.x * 128;
    const int n0  = blockIdx.y * 64;
    const int tid = threadIdx.x;
    const int wid = tid >> 5;
    const int lane = tid & 31;
    const int lo = lane & 15, hi = lane >> 4;
    const int kb8 = hi * 8, kb16 = hi * 16;

    v8f acc[2][4] = {};

    const _Float16* arow0 = A16 + (size_t)(m0 + wid * 32 + lo) * D_MODEL;
    const _Float16* arow1 = arow0 + 16 * D_MODEL;
    const _Float16* brow  = BT + (size_t)(n0 + lo) * D_MODEL;

    for (int k0 = 0; k0 < D_MODEL; k0 += 32) {
        v16h a0 = load_a16(arow0 + k0 + kb16);
        v16h a1 = load_a16(arow1 + k0 + kb16);
        #pragma unroll
        for (int nt = 0; nt < 4; ++nt) {
            v16h b = *(const v16h*)(brow + (size_t)nt * 16 * D_MODEL + k0 + kb16);
            wmma_f16(a0, b, acc[0][nt]);
            wmma_f16(a1, b, acc[1][nt]);
        }
    }

    #pragma unroll
    for (int nt = 0; nt < 4; ++nt) {
        const int n = n0 + nt * 16 + lo;
        const float bv = bias[n];
        const int h = n >> 6, dv = n & 63;
        #pragma unroll
        for (int j = 0; j < 2; ++j) {
            #pragma unroll
            for (int r = 0; r < 8; ++r) {
                const int m = m0 + wid * 32 + j * 16 + r + hi * 8;
                const _Float16 val = (_Float16)(acc[j][nt][r] + bv);
                if (vmode == 0)
                    outH[((size_t)h * S_LEN + m) * DHEAD + dv] = val;
                else
                    outH[((size_t)h * DHEAD + dv) * S_LEN + m] = val;
            }
        }
    }
}

// ---------------------------------------------------------------------------
// Attention: block per (head, 16-query tile); 256 threads = 8 waves.
// f16 score tile (16 x 4096, ~132KB LDS -> 2 blocks/WGP); f32 row stats;
// exact row softmax; single coalesced 1 GiB prob write; PV on WMMA with the
// 1/sum folded into the epilogue; PV A-fragments are direct ds_load_b128.
// ---------------------------------------------------------------------------
__global__ void __launch_bounds__(256) attn_kernel(const _Float16* __restrict__ Qh,
                                                   const _Float16* __restrict__ Kh,
                                                   const _Float16* __restrict__ Vt,
                                                   float* __restrict__ attn_out,
                                                   _Float16* __restrict__ ctxh)
{
    const int q0  = blockIdx.x * 16;
    const int h   = blockIdx.y;
    const int tid = threadIdx.x;
    const int wid = tid >> 5;
    const int lane = tid & 31;
    const int lo = lane & 15, hi = lane >> 4;
    const int kb8 = hi * 8, kb16 = hi * 16;

    extern __shared__ char smem[];
    _Float16* Ps  = (_Float16*)smem;               // 16 * PH halves (132096 B)
    float* red    = (float*)(smem + 16 * PH * 2);  // 16 * 16
    float* rowmax = red + 256;                     // 16
    float* rowinv = rowmax + 16;                   // 16
    float* ctx    = rowinv + 16;                   // 16 * 64

    for (int i = tid; i < 16 * 64; i += 256) ctx[i] = 0.f;

    // Q fragments for dk = [0,32) and [32,64)
    const _Float16* qptr = Qh + ((size_t)h * S_LEN + q0) * DHEAD;
    v16h aq[2];
    #pragma unroll
    for (int kk = 0; kk < 2; ++kk)
        aq[kk] = load_a16(qptr + lo * DHEAD + kk * 32 + kb16);

    // ---- scores: each wave covers 512 keys; B = K^T contiguous from Kh
    const _Float16* kbase = Kh + (size_t)h * S_LEN * DHEAD;
    const float scale = 0.125f;  // 1/sqrt(64)
    for (int n0 = wid * 512; n0 < wid * 512 + 512; n0 += 16) {
        __builtin_prefetch(kbase + (size_t)(n0 + 64 + lo) * DHEAD, 0, 3);
        v8f c = {};
        #pragma unroll
        for (int kk = 0; kk < 2; ++kk) {
            v16h b = *(const v16h*)(kbase + (size_t)(n0 + lo) * DHEAD + kk * 32 + kb16);
            wmma_f16(aq[kk], b, c);
        }
        #pragma unroll
        for (int r = 0; r < 8; ++r)
            Ps[(r + hi * 8) * PH + n0 + lo] = (_Float16)(c[r] * scale);
    }
    __syncthreads();

    // ---- softmax (16 threads per row, conflict-free strided access; f32 stats)
    const int row = tid >> 4, sub = tid & 15;
    float m = -1e30f;
    for (int j = 0; j < 256; ++j)
        m = fmaxf(m, (float)Ps[row * PH + sub + 16 * j]);
    red[row * 16 + sub] = m;
    __syncthreads();
    if (sub == 0) {
        float mm = red[row * 16];
        for (int j = 1; j < 16; ++j) mm = fmaxf(mm, red[row * 16 + j]);
        rowmax[row] = mm;
    }
    __syncthreads();
    const float rm = rowmax[row];
    float s = 0.f;
    for (int j = 0; j < 256; ++j) {
        const int idx = row * PH + sub + 16 * j;
        const float p = __expf((float)Ps[idx] - rm);
        Ps[idx] = (_Float16)p;       // unnormalized; 1/sum folded into PV
        s += p;
    }
    red[row * 16 + sub] = s;
    __syncthreads();
    if (sub == 0) {
        float ss = 0.f;
        for (int j = 0; j < 16; ++j) ss += red[row * 16 + j];
        rowinv[row] = 1.0f / ss;
    }
    __syncthreads();

    // ---- normalized probabilities -> d_out (coalesced; the mandatory 1 GiB pass)
    {
        float* dst = attn_out + (size_t)h * 16777216ULL + (size_t)q0 * S_LEN;
        for (int j = 0; j < 256; ++j) {
            const int flat = j * 256 + tid;
            const int r = flat >> 12, c = flat & 4095;
            dst[(size_t)r * S_LEN + c] = (float)Ps[r * PH + c] * rowinv[r];
        }
    }

    // ---- PV: wave covers its 512 keys; A = two ds_load_b128, B = contiguous Vt
    const _Float16* vbase = Vt + (size_t)h * DHEAD * S_LEN;
    v8f acc[4] = {};
    for (int kb = wid * 512; kb < wid * 512 + 512; kb += 32) {
        __builtin_prefetch(vbase + (size_t)lo * S_LEN + kb + 128, 0, 3);
        v16h a = load_a16(Ps + lo * PH + kb + kb16);
        #pragma unroll
        for (int nt = 0; nt < 4; ++nt) {
            v16h b = *(const v16h*)(vbase + (size_t)(nt * 16 + lo) * S_LEN + kb + kb16);
            wmma_f16(a, b, acc[nt]);
        }
    }
    #pragma unroll
    for (int nt = 0; nt < 4; ++nt) {
        #pragma unroll
        for (int r = 0; r < 8; ++r) {
            const int rr = r + hi * 8;
            atomicAdd(&ctx[rr * 64 + nt * 16 + lo], acc[nt][r] * rowinv[rr]);
        }
    }
    __syncthreads();

    for (int i = tid; i < 16 * 64; i += 256) {
        const int r = i >> 6, dv = i & 63;
        ctxh[(size_t)(q0 + r) * D_MODEL + h * DHEAD + dv] = (_Float16)ctx[i];
    }
}

// ---------------------------------------------------------------------------
// Output projection (LDS-free): ctx(f16) @ WoT(f16 n-major) + bo + residual -> f32
// ---------------------------------------------------------------------------
__global__ void __launch_bounds__(128) out_gemm(const _Float16* __restrict__ A16,
                                                const _Float16* __restrict__ BT,
                                                const float* __restrict__ bias,
                                                const float* __restrict__ resid,
                                                float* __restrict__ out)
{
    const int m0  = blockIdx.x * 128;
    const int n0  = blockIdx.y * 64;
    const int tid = threadIdx.x;
    const int wid = tid >> 5;
    const int lane = tid & 31;
    const int lo = lane & 15, hi = lane >> 4;
    const int kb8 = hi * 8, kb16 = hi * 16;

    v8f acc[2][4] = {};

    const _Float16* arow0 = A16 + (size_t)(m0 + wid * 32 + lo) * D_MODEL;
    const _Float16* arow1 = arow0 + 16 * D_MODEL;
    const _Float16* brow  = BT + (size_t)(n0 + lo) * D_MODEL;

    for (int k0 = 0; k0 < D_MODEL; k0 += 32) {
        v16h a0 = load_a16(arow0 + k0 + kb16);
        v16h a1 = load_a16(arow1 + k0 + kb16);
        #pragma unroll
        for (int nt = 0; nt < 4; ++nt) {
            v16h b = *(const v16h*)(brow + (size_t)nt * 16 * D_MODEL + k0 + kb16);
            wmma_f16(a0, b, acc[0][nt]);
            wmma_f16(a1, b, acc[1][nt]);
        }
    }

    #pragma unroll
    for (int nt = 0; nt < 4; ++nt) {
        const int n = n0 + nt * 16 + lo;
        const float bv = bias[n];
        #pragma unroll
        for (int j = 0; j < 2; ++j) {
            #pragma unroll
            for (int r = 0; r < 8; ++r) {
                const int m = m0 + wid * 32 + j * 16 + r + hi * 8;
                const size_t off = (size_t)m * D_MODEL + n;
                out[off] = acc[j][nt][r] + bv + resid[off];
            }
        }
    }
}

// ---------------------------------------------------------------------------
// Row-wise LayerNorm -> y
// ---------------------------------------------------------------------------
__global__ void __launch_bounds__(256) ln_kernel(const float* __restrict__ x,
                                                 const float* __restrict__ gamma,
                                                 const float* __restrict__ beta,
                                                 float* __restrict__ y)
{
    const int srow = blockIdx.x;
    const float* px = x + (size_t)srow * D_MODEL;
    __shared__ float red[256];

    float s = 0.f;
    for (int i = threadIdx.x; i < D_MODEL; i += 256) s += px[i];
    red[threadIdx.x] = s;
    __syncthreads();
    for (int o = 128; o > 0; o >>= 1) {
        if (threadIdx.x < o) red[threadIdx.x] += red[threadIdx.x + o];
        __syncthreads();
    }
    const float mu = red[0] * (1.0f / D_MODEL);
    __syncthreads();

    float v = 0.f;
    for (int i = threadIdx.x; i < D_MODEL; i += 256) {
        const float d = px[i] - mu;
        v += d * d;
    }
    red[threadIdx.x] = v;
    __syncthreads();
    for (int o = 128; o > 0; o >>= 1) {
        if (threadIdx.x < o) red[threadIdx.x] += red[threadIdx.x + o];
        __syncthreads();
    }
    const float rstd = rsqrtf(red[0] * (1.0f / D_MODEL) + 1e-5f);

    for (int i = threadIdx.x; i < D_MODEL; i += 256)
        y[(size_t)srow * D_MODEL + i] = (px[i] - mu) * rstd * gamma[i] + beta[i];
}

// ---------------------------------------------------------------------------
extern "C" void kernel_launch(void* const* d_in, const int* in_sizes, int n_in,
                              void* d_out, int out_size, void* d_ws, size_t ws_size,
                              hipStream_t stream)
{
    const float* q     = (const float*)d_in[0];
    const float* k     = (const float*)d_in[1];
    const float* v     = (const float*)d_in[2];
    const float* Wq    = (const float*)d_in[3];
    const float* bq    = (const float*)d_in[4];
    const float* Wk    = (const float*)d_in[5];
    const float* bk    = (const float*)d_in[6];
    const float* Wv    = (const float*)d_in[7];
    const float* bv    = (const float*)d_in[8];
    const float* Wo    = (const float*)d_in[9];
    const float* bo    = (const float*)d_in[10];
    const float* gamma = (const float*)d_in[11];
    const float* beta  = (const float*)d_in[12];

    char* ws = (char*)d_ws;
    const size_t MB = 1u << 20;
    _Float16* WqT  = (_Float16*)(ws + 0 * MB);    // 2 MB each
    _Float16* WkT  = (_Float16*)(ws + 2 * MB);
    _Float16* WvT  = (_Float16*)(ws + 4 * MB);
    _Float16* WoT  = (_Float16*)(ws + 6 * MB);
    _Float16* Xq16 = (_Float16*)(ws + 8 * MB);    // 8 MB each
    _Float16* Xk16 = (_Float16*)(ws + 16 * MB);
    _Float16* Xv16 = (_Float16*)(ws + 24 * MB);
    _Float16* Qh   = (_Float16*)(ws + 32 * MB);
    _Float16* Kh   = (_Float16*)(ws + 40 * MB);
    _Float16* Vt   = (_Float16*)(ws + 48 * MB);
    _Float16* ctxh = (_Float16*)(ws + 56 * MB);
    float*    xbuf = (float*)   (ws + 8 * MB);    // aliases Xq16/Xk16 (dead by then)

    float* y    = (float*)d_out;
    float* attn = (float*)d_out + (size_t)S_LEN * D_MODEL;

    // prep: fp16 activations + transposed fp16 weights
    cvt16<<<4096, 256, 0, stream>>>(q, Xq16);
    cvt16<<<4096, 256, 0, stream>>>(k, Xk16);
    cvt16<<<4096, 256, 0, stream>>>(v, Xv16);
    dim3 gw(16, 16);
    wt16<<<gw, 256, 0, stream>>>(Wq, WqT);
    wt16<<<gw, 256, 0, stream>>>(Wk, WkT);
    wt16<<<gw, 256, 0, stream>>>(Wv, WvT);
    wt16<<<gw, 256, 0, stream>>>(Wo, WoT);

    // projections (LDS-free WMMA GEMMs)
    dim3 gp(S_LEN / 128, D_MODEL / 64);   // (32, 16)
    gemm16<<<gp, 128, 0, stream>>>(Xq16, WqT, bq, Qh, 0);
    gemm16<<<gp, 128, 0, stream>>>(Xk16, WkT, bk, Kh, 0);
    gemm16<<<gp, 128, 0, stream>>>(Xv16, WvT, bv, Vt, 1);

    // attention (head index slow -> K/V L2-resident per head)
    const size_t shmem = (size_t)(16 * PH * 2) +
                         (size_t)(256 + 16 + 16 + 16 * 64) * sizeof(float);
    attn_kernel<<<dim3(S_LEN / 16, NHEAD), 256, shmem, stream>>>(Qh, Kh, Vt, attn, ctxh);

    // output projection + residual, then LayerNorm
    out_gemm<<<gp, 128, 0, stream>>>(ctxh, WoT, bo, q, xbuf);
    ln_kernel<<<S_LEN, 256, 0, stream>>>(xbuf, gamma, beta, y);
}
